// ViroShieldGenerator_8211977470513
// MI455X (gfx1250) — matrix operand order repaired and checked by reading, not compiled, not run on hardware
//
#include <hip/hip_runtime.h>
#include <hip/hip_bf16.h>
#include <math.h>

typedef __attribute__((ext_vector_type(16))) _Float16 v16h;
typedef __attribute__((ext_vector_type(8)))  _Float16 h8t;
typedef __attribute__((ext_vector_type(8)))  float    v8f;
typedef __attribute__((ext_vector_type(4)))  int      v4i;

#if defined(__has_builtin)
#  if __has_builtin(__builtin_amdgcn_global_load_async_to_lds_b128)
#    define USE_ASYNC_LDS 1
#  endif
#endif
#ifndef USE_ASYNC_LDS
#  define USE_ASYNC_LDS 0
#endif

#if USE_ASYNC_LDS
typedef __attribute__((address_space(1))) v4i gv4i;   // global int4
typedef __attribute__((address_space(3))) v4i lv4i;   // LDS int4
#endif

// ---------- helpers ----------
__device__ __forceinline__ int f2ord(float f) {
  int i = __float_as_int(f);
  return (i >= 0) ? i : (int)(i ^ 0x7fffffff);
}
__device__ __forceinline__ float ord2f(int i) {
  return __int_as_float((i >= 0) ? i : (int)(i ^ 0x7fffffff));
}

// ---------- embedding: x = emb[idx], fp32 + f16, zero pad rows ----------
__global__ void k_embed(const int* __restrict__ px, const float* __restrict__ emb,
                        float* __restrict__ xf, _Float16* __restrict__ xh,
                        int N, int Npad) {
  int i = blockIdx.x * blockDim.x + threadIdx.x;
  if (i >= Npad * 128) return;
  int row = i >> 7, c = i & 127;
  float v = 0.0f;
  if (row < N) v = emb[px[row] * 128 + c];
  xf[i] = v;
  xh[i] = (_Float16)v;
}

// ---------- f32 -> f16 convert ----------
__global__ void k_cvt16(const float* __restrict__ src, _Float16* __restrict__ dst, int n) {
  int i = blockIdx.x * blockDim.x + threadIdx.x;
  if (i < n) dst[i] = (_Float16)src[i];
}

// ---------- f32 [K,N] -> f16 [N,K] transpose-convert ----------
__global__ void k_cvt16_t(const float* __restrict__ src, _Float16* __restrict__ dst,
                          int K, int N) {
  int i = blockIdx.x * blockDim.x + threadIdx.x;
  if (i >= K * N) return;
  int k = i / N, n = i % N;
  dst[n * K + k] = (_Float16)src[i];
}

__global__ void k_zero(float* __restrict__ p, long n) {
  long i = (long)blockIdx.x * blockDim.x + threadIdx.x;
  if (i < n) p[i] = 0.f;
}

__global__ void k_init_seg(int* __restrict__ maxb, float* __restrict__ denom, int n) {
  int i = blockIdx.x * blockDim.x + threadIdx.x;
  if (i >= n) return;
  maxb[i] = (int)0x80000000;   // < mapped(-inf)
  denom[i] = 0.f;
}

// ---------- WMMA GEMM: C[M,N] = A[M,K] * Bt[N,K]^T + bias[N] ----------
// A,Bt f16 row-major, M%16==0, K in {32..256, %32==0}.
// Block = 8 waves; one 16-row A tile staged in LDS (async copy when available),
// each wave computes a 16x64 strip (4 N-tiles) = 16 chained v_wmma per wave.
__global__ void __launch_bounds__(256)
k_gemm(const _Float16* __restrict__ A, const _Float16* __restrict__ Bt,
       const float* __restrict__ bias, float* __restrict__ C,
       int M, int Nc, int K) {
  __shared__ __align__(16) _Float16 As[16 * 256];   // up to K=256 (8 KB)
  int tid  = threadIdx.x;
  int wave = tid >> 5, lane = tid & 31;
  int tm   = blockIdx.x;

  // ---- stage A row-tile (16 x K f16) into LDS, 16B per thread-chunk ----
  {
    const _Float16* ag = A + (size_t)tm * 16 * K;
    int chunks = 2 * K;                       // (16*K halves)/8 per 16B chunk
    for (int c = tid; c < chunks; c += 256) {
#if USE_ASYNC_LDS
      __builtin_amdgcn_global_load_async_to_lds_b128(
          (gv4i*)(ag + c * 8), (lv4i*)(As + c * 8), 0, 0);
#else
      ((uint4*)As)[c] = ((const uint4*)ag)[c];
#endif
    }
#if USE_ASYNC_LDS
#  if __has_builtin(__builtin_amdgcn_s_wait_asynccnt)
    __builtin_amdgcn_s_wait_asynccnt(0);
#  else
    asm volatile("s_wait_asynccnt 0" ::: "memory");
#  endif
#endif
  }
  __syncthreads();

  int tn0 = (blockIdx.y * 8 + wave) * 4;     // first of 4 N-tiles for this wave
  if (tn0 * 16 < Nc) {
    int lh = lane >> 4;                      // lane half
    int r  = lane & 15;
    const _Float16* arow = As + r * K;       // A: row M=r (from LDS)
    const _Float16* bp[4];
#pragma unroll
    for (int j = 0; j < 4; ++j)
      bp[j] = Bt + (size_t)((tn0 + j) * 16 + r) * K;   // B: col N=r of tile j
    v8f acc[4] = {};
    for (int k0 = 0; k0 < K; k0 += 32) {
      // A layout (16-bit 16x32): contiguous K range [lh*16, lh*16+16)
      h8t a0 = *(const h8t*)(arow + k0 + lh * 16);
      h8t a1 = *(const h8t*)(arow + k0 + lh * 16 + 8);
      v16h a;
#pragma unroll
      for (int i = 0; i < 8; ++i) { a[i] = a0[i]; a[8 + i] = a1[i]; }
#pragma unroll
      for (int j = 0; j < 4; ++j) {
        // B layout (16-bit 32x16): contiguous K range [k0+lh*16, k0+lh*16+16)
        v16h b = *(const v16h*)(bp[j] + k0 + lh * 16);
        acc[j] = __builtin_amdgcn_wmma_f32_16x16x32_f16(false, a, false, b, (short)0,
                                                        acc[j], false, false);
      }
    }
#pragma unroll
    for (int j = 0; j < 4; ++j) {
      int n = (tn0 + j) * 16 + r;
      float bv = bias ? bias[n] : 0.0f;
#pragma unroll
      for (int v = 0; v < 8; ++v) {          // C/D: VGPR v -> M = v + lh*8
        int mrow = tm * 16 + v + lh * 8;
        C[(size_t)mrow * Nc + n] = acc[j][v] + bv;
      }
    }
  }
}

// ---------- GATv2 edge logits + segment max (warp per (edge,head)) ----------
__global__ void __launch_bounds__(256)
k_edge_logit_max(const float* __restrict__ xl, const float* __restrict__ xr,
                 const int* __restrict__ ei, int E, int Ep,
                 const float* __restrict__ att,
                 float* __restrict__ z, int* __restrict__ maxb) {
  int wid = (blockIdx.x * blockDim.x + threadIdx.x) >> 5;
  int lane = threadIdx.x & 31;
  if (wid >= Ep * 4) return;
  int e = wid >> 2, h = wid & 3;
  int s, d;
  if (e < E) { s = ei[e]; d = ei[E + e]; } else { s = d = e - E; }
  const float* pl = xl + (size_t)s * 512 + h * 128;
  const float* pr = xr + (size_t)d * 512 + h * 128;
  const float* pa = att + h * 128;
  float sum = 0.f;
  for (int c = lane; c < 128; c += 32) {
    float v = pl[c] + pr[c];
    v = v > 0.f ? v : 0.2f * v;            // leaky_relu(0.2)
    sum += v * pa[c];
  }
#pragma unroll
  for (int off = 16; off > 0; off >>= 1) sum += __shfl_xor(sum, off, 32);
  if (lane == 0) {
    z[e * 4 + h] = sum;
    atomicMax(&maxb[d * 4 + h], f2ord(sum));
  }
}

// ---------- exp(logit - max) + segment sum ----------
__global__ void k_z_denom(float* __restrict__ z, const int* __restrict__ maxb,
                          float* __restrict__ denom, const int* __restrict__ ei,
                          int E, int Ep) {
  int i = blockIdx.x * blockDim.x + threadIdx.x;
  if (i >= Ep * 4) return;
  int e = i >> 2, h = i & 3;
  int d = (e < E) ? ei[E + e] : (e - E);
  float m = ord2f(maxb[d * 4 + h]);
  float zz = __expf(z[i] - m);
  z[i] = zz;
  atomicAdd(&denom[d * 4 + h], zz);
}

// ---------- weighted scatter-add: acc[dst] += alpha * xl[src] ----------
__global__ void __launch_bounds__(256)
k_aggregate(const float* __restrict__ xl, const float* __restrict__ z,
            const float* __restrict__ denom, const int* __restrict__ ei,
            int E, float* __restrict__ acc) {
  int e = blockIdx.x;
  int s, d;
  if (e < E) { s = ei[e]; d = ei[E + e]; } else { s = d = e - E; }
  for (int c = threadIdx.x; c < 512; c += 256) {
    int h = c >> 7;
    float alpha = z[e * 4 + h] / denom[d * 4 + h];
    atomicAdd(&acc[(size_t)d * 512 + c], alpha * xl[(size_t)s * 512 + c]);
  }
}

// ---------- mean over 4 heads + bias ----------
__global__ void k_headmean(const float* __restrict__ acc, const float* __restrict__ gb,
                           float* __restrict__ gout, int N) {
  int i = blockIdx.x * blockDim.x + threadIdx.x;
  if (i >= N * 128) return;
  int n = i >> 7, c = i & 127;
  const float* p = acc + (size_t)n * 512 + c;
  gout[i] = 0.25f * (p[0] + p[128] + p[256] + p[384]) + gb[c];
}

// ---------- BN training stats: one block per channel ----------
__global__ void __launch_bounds__(256)
k_bnstats(const float* __restrict__ x, float* __restrict__ stats, int N) {
  int c = blockIdx.x;
  float s = 0.f, s2 = 0.f;
  for (int n = threadIdx.x; n < N; n += 256) {
    float v = x[(size_t)n * 128 + c];
    s += v; s2 += v * v;
  }
  __shared__ float ss[256], sq[256];
  ss[threadIdx.x] = s; sq[threadIdx.x] = s2;
  __syncthreads();
  for (int st = 128; st > 0; st >>= 1) {
    if (threadIdx.x < st) { ss[threadIdx.x] += ss[threadIdx.x + st];
                            sq[threadIdx.x] += sq[threadIdx.x + st]; }
    __syncthreads();
  }
  if (threadIdx.x == 0) {
    float mu = ss[0] / N;
    stats[c] = mu;
    stats[128 + c] = sq[0] / N - mu * mu;   // biased variance
  }
}

// ---------- BN apply + ELU + residual, refresh f16 copy ----------
__global__ void k_bnapply(const float* __restrict__ gout, const float* __restrict__ stats,
                          const float* __restrict__ g, const float* __restrict__ b,
                          float* __restrict__ xf, _Float16* __restrict__ xh, int N) {
  int i = blockIdx.x * blockDim.x + threadIdx.x;
  if (i >= N * 128) return;
  int c = i & 127;
  float y = (gout[i] - stats[c]) * rsqrtf(stats[128 + c] + 1e-5f) * g[c] + b[c];
  float e = y > 0.f ? y : expm1f(y);        // elu
  float nx = e + xf[i];
  xf[i] = nx;
  xh[i] = (_Float16)nx;
}

// ---------- ligand self-attention (4 heads, hd=32, L<=64): thread per (l,h) ----------
__global__ void k_sa_attn(const float* __restrict__ qkv, float* __restrict__ O, int ML) {
  int t = blockIdx.x * blockDim.x + threadIdx.x;
  if (t >= ML * 4) return;
  int l = t >> 2, h = t & 3;
  int cap = ML < 64 ? ML : 64;
  const float scale = 0.17677669529663687f;  // 1/sqrt(32)
  float sc[64];
  float m = -INFINITY;
  const float* qp = qkv + l * 384 + h * 32;
  for (int s = 0; s < cap; ++s) {
    const float* kp = qkv + s * 384 + 128 + h * 32;
    float d0 = 0.f;
    for (int d = 0; d < 32; ++d) d0 += qp[d] * kp[d];
    d0 *= scale;
    sc[s] = d0;
    m = fmaxf(m, d0);
  }
  float Z = 0.f;
  for (int s = 0; s < cap; ++s) { sc[s] = __expf(sc[s] - m); Z += sc[s]; }
  float inv = 1.f / Z;
  for (int d = 0; d < 32; ++d) {
    float o = 0.f;
    for (int s = 0; s < cap; ++s) o += sc[s] * qkv[s * 384 + 256 + h * 32 + d];
    O[l * 128 + h * 32 + d] = o * inv;
  }
}

// ---------- cross-attention flash pass (8 heads, hd=16): block per (l,h) ----------
__global__ void __launch_bounds__(256)
k_ca_attn(const float* __restrict__ Q, const float* __restrict__ KV,
          float* __restrict__ O, int Np) {
  int l = blockIdx.x, h = blockIdx.y;
  int tid = threadIdx.x;
  float q[16];
#pragma unroll
  for (int d = 0; d < 16; ++d) q[d] = Q[l * 128 + h * 16 + d];
  float m = -INFINITY, Z = 0.f, acc[16];
#pragma unroll
  for (int d = 0; d < 16; ++d) acc[d] = 0.f;
  for (int s = tid; s < Np; s += 256) {
    if (s + 256 < Np)                                  // hint next KV line into cache
      __builtin_prefetch(KV + (size_t)(s + 256) * 256 + h * 16, 0, 0);
    const float* kp = KV + (size_t)s * 256 + h * 16;   // cols 0..127 = K
    float sc = 0.f;
#pragma unroll
    for (int d = 0; d < 16; ++d) sc += q[d] * kp[d];
    sc *= 0.25f;                                       // 1/sqrt(16)
    if (sc > m) {
      float rr = __expf(m - sc);
      Z *= rr;
#pragma unroll
      for (int d = 0; d < 16; ++d) acc[d] *= rr;
      m = sc;
    }
    float p = __expf(sc - m);
    Z += p;
    const float* vp = kp + 128;                        // cols 128..255 = V
#pragma unroll
    for (int d = 0; d < 16; ++d) acc[d] += p * vp[d];
  }
  __shared__ float shm[256], shz[256], sha[256 * 16];
  shm[tid] = m; shz[tid] = Z;
#pragma unroll
  for (int d = 0; d < 16; ++d) sha[tid * 16 + d] = acc[d];
  __syncthreads();
  if (tid == 0) {
    float gm = shm[0], gz = shz[0], ga[16];
    for (int d = 0; d < 16; ++d) ga[d] = sha[d];
    for (int t = 1; t < 256; ++t) {
      if (shz[t] == 0.f) continue;
      float mt = shm[t];
      float nm = fmaxf(gm, mt);
      float r1 = __expf(gm - nm), r2 = __expf(mt - nm);
      gz = gz * r1 + shz[t] * r2;
      for (int d = 0; d < 16; ++d) ga[d] = ga[d] * r1 + sha[t * 16 + d] * r2;
      gm = nm;
    }
    float inv = 1.f / gz;
    for (int d = 0; d < 16; ++d) O[l * 128 + h * 16 + d] = ga[d] * inv;
  }
}

// ---------- out = LN(a+b)*g+be, fp32 + f16; thread per row ----------
__global__ void k_add_ln(const float* __restrict__ a, const float* __restrict__ b,
                         const float* __restrict__ g, const float* __restrict__ be,
                         float* __restrict__ outf, _Float16* __restrict__ outh, int rows) {
  int r = blockIdx.x * blockDim.x + threadIdx.x;
  if (r >= rows) return;
  float v[128], mu = 0.f;
  for (int c = 0; c < 128; ++c) { v[c] = a[r * 128 + c] + b[r * 128 + c]; mu += v[c]; }
  mu *= (1.f / 128.f);
  float var = 0.f;
  for (int c = 0; c < 128; ++c) { float dd = v[c] - mu; var += dd * dd; }
  var *= (1.f / 128.f);
  float inv = rsqrtf(var + 1e-5f);
  for (int c = 0; c < 128; ++c) {
    float y = (v[c] - mu) * inv * g[c] + be[c];
    outf[r * 128 + c] = y;
    outh[r * 128 + c] = (_Float16)y;
  }
}

// ---------- silu, write fp32 + f16 ----------
__global__ void k_silu_cvt(const float* __restrict__ src, float* __restrict__ dstf,
                           _Float16* __restrict__ dsth, int n) {
  int i = blockIdx.x * blockDim.x + threadIdx.x;
  if (i >= n) return;
  float x = src[i];
  float y = x / (1.f + __expf(-x));
  dstf[i] = y;
  dsth[i] = (_Float16)y;
}

// ---------- tiny final layer (outd = 3 or 6), plain fp32 ----------
__global__ void k_head_out(const float* __restrict__ hsrc, const float* __restrict__ w3,
                           const float* __restrict__ b3, float* __restrict__ out,
                           int ML, int outd) {
  int t = blockIdx.x * blockDim.x + threadIdx.x;
  if (t >= ML * outd) return;
  int l = t / outd, j = t % outd;
  float s = b3[j];
  for (int k = 0; k < 128; ++k) s += hsrc[l * 128 + k] * w3[k * outd + j];
  out[t] = s;
}

// =====================================================================
extern "C" void kernel_launch(void* const* d_in, const int* in_sizes, int n_in,
                              void* d_out, int out_size, void* d_ws, size_t ws_size,
                              hipStream_t stream) {
  (void)n_in; (void)out_size; (void)ws_size;
  const int N    = in_sizes[0];
  const int E    = in_sizes[1] / 2;
  const int ML   = in_sizes[2] / 128;
  const int Ep   = E + N;
  const int Npad = (N + 15) & ~15;

  const int* px = (const int*)d_in[0];
  const int* ei = (const int*)d_in[1];
  const float* seed = (const float*)d_in[2];
  auto F = [&](int i) { return (const float*)d_in[i]; };
  // param indices: 3 emb; 4-9 g1(wl,bl,wr,br,att,b); 10-15 g2; 16-19 bn; 20-23 sa;
  // 24-27 ca; 28-31 ln; 32-37 c-head; 38-43 t-head

  char* wp = (char*)d_ws;
  auto alloc = [&](size_t bytes) -> void* {
    void* p = (void*)wp;
    wp += (bytes + 255) & ~(size_t)255;
    return p;
  };
  float*    xf    = (float*)alloc((size_t)Npad * 128 * 4);
  _Float16* xh    = (_Float16*)alloc((size_t)Npad * 128 * 2);
  float*    gout  = (float*)alloc((size_t)Npad * 128 * 4);
  float*    big1  = (float*)alloc((size_t)Npad * 512 * 4);   // xl / later CA KV
  float*    big2  = (float*)alloc((size_t)Npad * 512 * 4);   // xr / later agg acc
  float*    zb    = (float*)alloc((size_t)Ep * 4 * 4);
  int*      maxb  = (int*)alloc((size_t)N * 4 * 4);
  float*    denom = (float*)alloc((size_t)N * 4 * 4);
  float*    stats = (float*)alloc(256 * 4);
  _Float16* wT1   = (_Float16*)alloc(512 * 128 * 2);
  _Float16* wT2   = (_Float16*)alloc(512 * 128 * 2);
  _Float16* wsa   = (_Float16*)alloc(384 * 128 * 2);
  _Float16* wca   = (_Float16*)alloc(384 * 128 * 2);
  _Float16* wo16  = (_Float16*)alloc(128 * 128 * 2);
  _Float16* seed16= (_Float16*)alloc((size_t)ML * 128 * 2);
  float*    qkv   = (float*)alloc((size_t)ML * 384 * 4);
  float*    osa   = (float*)alloc((size_t)ML * 128 * 4);
  _Float16* o16   = (_Float16*)alloc((size_t)ML * 128 * 2);
  float*    selfo = (float*)alloc((size_t)ML * 128 * 4);
  float*    lig   = (float*)alloc((size_t)ML * 128 * 4);
  _Float16* lig16 = (_Float16*)alloc((size_t)ML * 128 * 2);
  float*    Qca   = (float*)alloc((size_t)ML * 128 * 4);
  float*    oca   = (float*)alloc((size_t)ML * 128 * 4);
  float*    fin   = (float*)alloc((size_t)ML * 128 * 4);
  _Float16* fin16 = (_Float16*)alloc((size_t)ML * 128 * 2);
  float*    h1    = (float*)alloc((size_t)ML * 256 * 4);
  _Float16* h116  = (_Float16*)alloc((size_t)ML * 256 * 2);
  float*    h2p   = (float*)alloc((size_t)ML * 128 * 4);
  float*    h2    = (float*)alloc((size_t)ML * 128 * 4);
  _Float16* h216  = (_Float16*)alloc((size_t)ML * 128 * 2);

  auto nb = [](long total, int b) { return (int)((total + b - 1) / b); };
  auto gemm = [&](const _Float16* A, const _Float16* Bt, const float* bias,
                  float* C, int M, int Nc, int K) {
    dim3 grid(M / 16, (Nc / 16 + 31) / 32);   // block covers 8 waves x 4 tiles = 512 cols
    k_gemm<<<grid, 256, 0, stream>>>(A, Bt, bias, C, M, Nc, K);
  };

  // ---- embedding ----
  k_embed<<<nb((long)Npad * 128, 256), 256, 0, stream>>>(px, F(3), xf, xh, N, Npad);

  // ---- two GATv2 layers ----
  for (int layer = 0; layer < 2; ++layer) {
    int pb = (layer == 0) ? 4 : 10;
    const float* wl = F(pb); const float* bl = F(pb + 1);
    const float* wr = F(pb + 2); const float* br = F(pb + 3);
    const float* att = F(pb + 4); const float* gb = F(pb + 5);
    const float* bng = F(16 + layer * 2); const float* bnb = F(17 + layer * 2);

    k_cvt16_t<<<nb(128 * 512, 256), 256, 0, stream>>>(wl, wT1, 128, 512);
    k_cvt16_t<<<nb(128 * 512, 256), 256, 0, stream>>>(wr, wT2, 128, 512);
    gemm(xh, wT1, bl, big1, Npad, 512, 128);                 // xl = x@wl+bl
    gemm(xh, wT2, br, big2, Npad, 512, 128);                 // xr = x@wr+br
    k_init_seg<<<nb((long)N * 4, 256), 256, 0, stream>>>(maxb, denom, N * 4);
    k_edge_logit_max<<<nb((long)Ep * 4 * 32, 256), 256, 0, stream>>>(
        big1, big2, ei, E, Ep, att, zb, maxb);
    k_z_denom<<<nb((long)Ep * 4, 256), 256, 0, stream>>>(zb, maxb, denom, ei, E, Ep);
    k_zero<<<nb((long)Npad * 512, 256), 256, 0, stream>>>(big2, (long)Npad * 512);
    k_aggregate<<<Ep, 256, 0, stream>>>(big1, zb, denom, ei, E, big2);
    k_headmean<<<nb((long)N * 128, 256), 256, 0, stream>>>(big2, gb, gout, N);
    k_bnstats<<<128, 256, 0, stream>>>(gout, stats, N);
    k_bnapply<<<nb((long)N * 128, 256), 256, 0, stream>>>(gout, stats, bng, bnb, xf, xh, N);
  }

  // ---- ligand self-attention ----
  k_cvt16<<<nb((long)ML * 128, 256), 256, 0, stream>>>(seed, seed16, ML * 128);
  k_cvt16<<<nb(384 * 128, 256), 256, 0, stream>>>(F(20), wsa, 384 * 128);
  gemm(seed16, wsa, F(21), qkv, ML, 384, 128);               // [Q|K|V]
  k_sa_attn<<<nb((long)ML * 4, 256), 256, 0, stream>>>(qkv, osa, ML);
  k_cvt16<<<nb((long)ML * 128, 256), 256, 0, stream>>>(osa, o16, ML * 128);
  k_cvt16<<<nb(128 * 128, 256), 256, 0, stream>>>(F(22), wo16, 128 * 128);
  gemm(o16, wo16, F(23), selfo, ML, 128, 128);               // out-proj
  k_add_ln<<<nb(ML, 64), 64, 0, stream>>>(seed, selfo, F(28), F(29), lig, lig16, ML);

  // ---- cross-attention ligand -> pocket ----
  k_cvt16<<<nb(384 * 128, 256), 256, 0, stream>>>(F(24), wca, 384 * 128);
  gemm(lig16, wca, F(25), Qca, ML, 128, 128);                // Q (w_in rows 0..127)
  gemm(xh, wca + 128 * 128, F(25) + 128, big1, Npad, 256, 128); // KV (rows 128..383)
  k_ca_attn<<<dim3(ML, 8), 256, 0, stream>>>(Qca, big1, oca, N);
  k_cvt16<<<nb((long)ML * 128, 256), 256, 0, stream>>>(oca, o16, ML * 128);
  k_cvt16<<<nb(128 * 128, 256), 256, 0, stream>>>(F(26), wo16, 128 * 128);
  gemm(o16, wo16, F(27), selfo, ML, 128, 128);               // out-proj
  k_add_ln<<<nb(ML, 64), 64, 0, stream>>>(selfo, lig, F(30), F(31), fin, fin16, ML);

  // ---- coord / type MLP heads ----
  float* out = (float*)d_out;
  for (int head = 0; head < 2; ++head) {
    int pb = (head == 0) ? 32 : 38;
    int outd = (head == 0) ? 3 : 6;
    float* outp = (head == 0) ? out : out + (size_t)ML * 3;
    k_cvt16_t<<<nb(128 * 256, 256), 256, 0, stream>>>(F(pb), wT1, 128, 256);
    gemm(fin16, wT1, F(pb + 1), h1, ML, 256, 128);
    k_silu_cvt<<<nb((long)ML * 256, 256), 256, 0, stream>>>(h1, h1, h116, ML * 256);
    k_cvt16_t<<<nb(256 * 128, 256), 256, 0, stream>>>(F(pb + 2), wT2, 256, 128);
    gemm(h116, wT2, F(pb + 3), h2p, ML, 128, 256);
    k_silu_cvt<<<nb((long)ML * 128, 256), 256, 0, stream>>>(h2p, h2, h216, ML * 128);
    k_head_out<<<nb((long)ML * outd, 256), 256, 0, stream>>>(h2, F(pb + 4), F(pb + 5),
                                                             outp, ML, outd);
  }
}
